// FeGAN_77292231458959
// MI455X (gfx1250) — matrix operand order repaired and checked by reading, not compiled, not run on hardware
//
#include <hip/hip_runtime.h>
#include <hip/hip_bf16.h>

#define HEADS 8
#define HID 32
#define GRAPHS 64
#define CLASSES 10
#define NEG_SLOPE 0.2f
#define TN 2   // N-tiles per wave in the WMMA GEMM (divides 16 and 2)

typedef __attribute__((ext_vector_type(16))) __bf16 v16bf;
typedef __attribute__((ext_vector_type(8)))  float  v8f;

__device__ __forceinline__ float lrelu(float x) { return x > 0.f ? x : NEG_SLOPE * x; }

__device__ __forceinline__ void atomicMaxF(float* addr, float val) {
    int* ia = (int*)addr;
    int old = *ia;
    while (val > __int_as_float(old)) {
        int assumed = old;
        old = atomicCAS(ia, assumed, __float_as_int(val));
        if (old == assumed) break;
    }
}

// ---------------- WMMA GEMM: Y[M,F] = X[M,K] @ W[K,F]  (bf16 in, f32 acc) ----
// One wave computes a 16 x (16*TN) strip: A registers converted once, reused
// for TN WMMAs. M,F multiples of 16; K multiple of 32; (F/16) % TN == 0.
__global__ void wmma_gemm_bf16(const float* __restrict__ X, const float* __restrict__ W,
                               float* __restrict__ Y, int M, int K, int F) {
    int lane = threadIdx.x & 31;
    long long grp = (long long)blockIdx.x * (blockDim.x >> 5) + (threadIdx.x >> 5);
    int tilesN = F >> 4;
    int grpN = tilesN / TN;
    long long grpTotal = (long long)(M >> 4) * grpN;
    if (grp >= grpTotal) return;            // wave-uniform exit (EXEC all-1 for WMMA)
    int m0 = (int)(grp / grpN) << 4;
    int n0 = (int)(grp % grpN) * (16 * TN);
    int g = lane >> 4;                      // half-wave group
    int r = lane & 15;

    v8f acc[TN];
#pragma unroll
    for (int j = 0; j < TN; ++j) acc[j] = (v8f){};

    for (int k0 = 0; k0 < K; k0 += 32) {
        // A 16x32 bf16 layout: lane row = m0+r, K = k0 + 8*g + 16*(e>>3) + (e&7)
        // -> per lane: two contiguous 8-float runs at +0 and +16.
        const float* xrow = X + (long long)(m0 + r) * K + k0 + 8 * g;
        const float4 a0 = *(const float4*)(xrow + 0);
        const float4 a1 = *(const float4*)(xrow + 4);
        const float4 a2 = *(const float4*)(xrow + 16);
        const float4 a3 = *(const float4*)(xrow + 20);
        v16bf a;
        a[0]  = (__bf16)a0.x; a[1]  = (__bf16)a0.y; a[2]  = (__bf16)a0.z; a[3]  = (__bf16)a0.w;
        a[4]  = (__bf16)a1.x; a[5]  = (__bf16)a1.y; a[6]  = (__bf16)a1.z; a[7]  = (__bf16)a1.w;
        a[8]  = (__bf16)a2.x; a[9]  = (__bf16)a2.y; a[10] = (__bf16)a2.z; a[11] = (__bf16)a2.w;
        a[12] = (__bf16)a3.x; a[13] = (__bf16)a3.y; a[14] = (__bf16)a3.z; a[15] = (__bf16)a3.w;

#pragma unroll
        for (int j = 0; j < TN; ++j) {
            // B 32x16 bf16 layout: lane col = n0+16j+r, K = k0 + 8*g + 16*(e>>3) + (e&7)
            const float* wc = W + (long long)(k0 + 8 * g) * F + n0 + 16 * j + r;
            v16bf b;
#pragma unroll
            for (int e = 0; e < 16; ++e) b[e] = (__bf16)wc[(long long)(e + 8 * (e >> 3)) * F];
            acc[j] = __builtin_amdgcn_wmma_f32_16x16x32_bf16(false, a, false, b,
                                                             (short)0, acc[j], false, false);
        }
    }
    // C/D layout: VGPR rr -> row m0 + rr + 8*g, col n0 + 16*j + r
#pragma unroll
    for (int j = 0; j < TN; ++j) {
        float* yt = Y + (long long)m0 * F + n0 + 16 * j;
#pragma unroll
        for (int rr = 0; rr < 8; ++rr)
            yt[(long long)(rr + 8 * g) * F + r] = acc[j][rr];
    }
}

// ---------------- per-node attention dot products ---------------------------
__global__ void attn_coef(const float* __restrict__ feat, const float* __restrict__ a_src,
                          const float* __restrict__ a_dst, float* __restrict__ alS,
                          float* __restrict__ alD, long long Nn, int H, int D) {
    long long t = (long long)blockIdx.x * blockDim.x + threadIdx.x;
    if (t >= Nn * H) return;
    int h = (int)(t % H);
    const float* f = feat + t * D;         // feat is [N, H*D]: row chunk (n*H+h)*D
    const float* as = a_src + h * D;
    const float* ad = a_dst + h * D;
    float s = 0.f, d = 0.f;
#pragma unroll 8
    for (int i = 0; i < D; ++i) { s += f[i] * as[i]; d += f[i] * ad[i]; }
    alS[t] = s; alD[t] = d;
}

// ---------------- softmax max: self-loop init + edge atomic max -------------
__global__ void self_max_k(const float* __restrict__ alS, const float* __restrict__ alD,
                           float* __restrict__ m, long long NH) {
    long long t = (long long)blockIdx.x * blockDim.x + threadIdx.x;
    if (t >= NH) return;
    m[t] = lrelu(alS[t] + alD[t]);
}

__global__ void edge_max_k(const long long* __restrict__ src, const long long* __restrict__ dst,
                           const float* __restrict__ alS, const float* __restrict__ alD,
                           float* __restrict__ m, long long E, int H) {
    long long t = (long long)blockIdx.x * blockDim.x + threadIdx.x;
    if (t >= E * H) return;
    long long e = t / H; int h = (int)(t % H);
    long long s = src[e], d = dst[e];
    float ev = lrelu(alS[s * H + h] + alD[d * H + h]);
    atomicMaxF(&m[d * H + h], ev);
}

// ---------------- softmax denom: self init + edge atomic add ----------------
__global__ void self_sum_k(const float* __restrict__ alS, const float* __restrict__ alD,
                           const float* __restrict__ m, float* __restrict__ z, long long NH) {
    long long t = (long long)blockIdx.x * blockDim.x + threadIdx.x;
    if (t >= NH) return;
    z[t] = expf(lrelu(alS[t] + alD[t]) - m[t]);
}

__global__ void edge_sum_k(const long long* __restrict__ src, const long long* __restrict__ dst,
                           const float* __restrict__ alS, const float* __restrict__ alD,
                           const float* __restrict__ m, float* __restrict__ z,
                           long long E, int H) {
    long long t = (long long)blockIdx.x * blockDim.x + threadIdx.x;
    if (t >= E * H) return;
    long long e = t / H; int h = (int)(t % H);
    long long s = src[e], d = dst[e];
    float ev = lrelu(alS[s * H + h] + alD[d * H + h]);
    atomicAdd(&z[d * H + h], expf(ev - m[d * H + h]));
}

// ---------------- aggregation: self init (plain store) + edge atomic add ----
__global__ void self_agg_k(const float* __restrict__ alS, const float* __restrict__ alD,
                           const float* __restrict__ m, const float* __restrict__ z,
                           const float* __restrict__ feat, float* __restrict__ agg,
                           long long Nn, int H, int D) {
    long long t = (long long)blockIdx.x * blockDim.x + threadIdx.x;  // over N*H*D
    if (t >= Nn * (long long)(H * D)) return;
    long long nh = t / D;                     // n*H + h
    float alpha = expf(lrelu(alS[nh] + alD[nh]) - m[nh]) / z[nh];
    agg[t] = feat[t] * alpha;                 // self-loop: src == dst, initializes agg
}

__global__ void edge_agg_k(const long long* __restrict__ src, const long long* __restrict__ dst,
                           const float* __restrict__ alS, const float* __restrict__ alD,
                           const float* __restrict__ m, const float* __restrict__ z,
                           const float* __restrict__ feat, float* __restrict__ agg,
                           long long E, int H, int D) {
    int F = H * D, cpe = F >> 2;              // float4 chunks per edge
    long long t = (long long)blockIdx.x * blockDim.x + threadIdx.x;
    if (t >= E * cpe) return;
    long long e = t / cpe; int c = (int)(t % cpe);
    int h = (c << 2) / D;
    long long s = src[e], d = dst[e];
    float ev = lrelu(alS[s * H + h] + alD[d * H + h]);
    float alpha = expf(ev - m[d * H + h]) / z[d * H + h];
    const float4 f = *(const float4*)(feat + s * (long long)F + (c << 2));
    float* o = agg + d * (long long)F + (c << 2);
    atomicAdd(o + 0, f.x * alpha);
    atomicAdd(o + 1, f.y * alpha);
    atomicAdd(o + 2, f.z * alpha);
    atomicAdd(o + 3, f.w * alpha);
}

// ---------------- bias + optional ELU (in-place capable) --------------------
__global__ void bias_elu_k(const float* __restrict__ agg, const float* __restrict__ b,
                           float* __restrict__ out, long long total, int F, int doElu) {
    long long t = (long long)blockIdx.x * blockDim.x + threadIdx.x;
    if (t >= total) return;
    float v = agg[t] + b[(int)(t % F)];
    if (doElu) v = v > 0.f ? v : (expf(v) - 1.f);
    out[t] = v;
}

// ---------------- pooling + classifier --------------------------------------
__global__ void fill_k(float* p, float v, long long n) {
    long long t = (long long)blockIdx.x * blockDim.x + threadIdx.x;
    if (t < n) p[t] = v;
}

__global__ void pool_k(const float* __restrict__ h3, const long long* __restrict__ batch,
                       float* __restrict__ sums, float* __restrict__ cnt, long long Nn) {
    long long t = (long long)blockIdx.x * blockDim.x + threadIdx.x;  // over N*32
    if (t >= Nn * HID) return;
    long long n = t / HID; int f = (int)(t % HID);
    long long b = batch[n];
    atomicAdd(&sums[b * HID + f], h3[t]);
    if (f == 0) atomicAdd(&cnt[b], 1.0f);
}

__global__ void classify_k(const float* __restrict__ sums, const float* __restrict__ cnt,
                           const float* __restrict__ Wc, const float* __restrict__ bc,
                           float* __restrict__ out) {
    int g = threadIdx.x;
    if (g >= GRAPHS) return;
    float c = fmaxf(cnt[g], 1.0f);
    float p[HID];
#pragma unroll
    for (int i = 0; i < HID; ++i) p[i] = sums[g * HID + i] / c;
    float lg[CLASSES], mx = -3.0e38f;
#pragma unroll
    for (int j = 0; j < CLASSES; ++j) {
        float a = bc[j];
#pragma unroll
        for (int i = 0; i < HID; ++i) a += p[i] * Wc[i * CLASSES + j];
        lg[j] = a; mx = fmaxf(mx, a);
    }
    float ssum = 0.f;
#pragma unroll
    for (int j = 0; j < CLASSES; ++j) ssum += expf(lg[j] - mx);
    float lse = mx + logf(ssum);
#pragma unroll
    for (int j = 0; j < CLASSES; ++j) out[g * CLASSES + j] = lg[j] - lse;
}

static inline unsigned gblk(long long total, int tpb) {
    return (unsigned)((total + tpb - 1) / tpb);
}

extern "C" void kernel_launch(void* const* d_in, const int* in_sizes, int n_in,
                              void* d_out, int out_size, void* d_ws, size_t ws_size,
                              hipStream_t stream) {
    const float*     x     = (const float*)d_in[0];
    const long long* ei    = (const long long*)d_in[1];
    const long long* batch = (const long long*)d_in[2];
    const float *W1 = (const float*)d_in[3],  *as1 = (const float*)d_in[4],
                *ad1 = (const float*)d_in[5], *b1 = (const float*)d_in[6];
    const float *W2 = (const float*)d_in[7],  *as2 = (const float*)d_in[8],
                *ad2 = (const float*)d_in[9], *b2 = (const float*)d_in[10];
    const float *W3 = (const float*)d_in[11], *as3 = (const float*)d_in[12],
                *ad3 = (const float*)d_in[13], *b3 = (const float*)d_in[14];
    const float *Wc = (const float*)d_in[15], *bc = (const float*)d_in[16];

    const long long Nn = in_sizes[0] / 128;   // 100000
    const long long Ee = in_sizes[1] / 2;     // 1600000
    const long long* src = ei;
    const long long* dst = ei + Ee;
    const int F = HEADS * HID;                // 256

    float* bufA = (float*)d_ws;               // activations in / agg out (N*256)
    float* bufB = bufA + (size_t)Nn * F;      // transformed features     (N*256)
    float* alS  = bufB + (size_t)Nn * F;      // (N*8)
    float* alD  = alS  + (size_t)Nn * HEADS;
    float* mb   = alD  + (size_t)Nn * HEADS;
    float* zb   = mb   + (size_t)Nn * HEADS;
    float* sums = zb   + (size_t)Nn * HEADS;  // (64*32)
    float* cnt  = sums + GRAPHS * HID;        // (64)

    const int T = 256;

    auto run_layer = [&](const float* in, int K, const float* W, const float* as_,
                         const float* ad_, const float* b_, int H, int D, int doElu) {
        int Fo = H * D;
        long long NH = Nn * H;
        long long grps = (Nn >> 4) * ((Fo >> 4) / TN);
        wmma_gemm_bf16<<<gblk(grps * 32, T), T, 0, stream>>>(in, W, bufB, (int)Nn, K, Fo);
        attn_coef<<<gblk(NH, T), T, 0, stream>>>(bufB, as_, ad_, alS, alD, Nn, H, D);
        self_max_k<<<gblk(NH, T), T, 0, stream>>>(alS, alD, mb, NH);
        edge_max_k<<<gblk(Ee * H, T), T, 0, stream>>>(src, dst, alS, alD, mb, Ee, H);
        self_sum_k<<<gblk(NH, T), T, 0, stream>>>(alS, alD, mb, zb, NH);
        edge_sum_k<<<gblk(Ee * H, T), T, 0, stream>>>(src, dst, alS, alD, mb, zb, Ee, H);
        self_agg_k<<<gblk(Nn * Fo, T), T, 0, stream>>>(alS, alD, mb, zb, bufB, bufA, Nn, H, D);
        edge_agg_k<<<gblk(Ee * (Fo >> 2), T), T, 0, stream>>>(src, dst, alS, alD, mb, zb,
                                                              bufB, bufA, Ee, H, D);
        bias_elu_k<<<gblk(Nn * Fo, T), T, 0, stream>>>(bufA, b_, bufA, Nn * (long long)Fo,
                                                       Fo, doElu);
    };

    // Layer 1: x[N,128] -> [N,256], ELU
    run_layer(x, 128, W1, as1, ad1, b1, HEADS, HID, 1);
    // Layer 2: [N,256] -> [N,256], ELU  (input = bufA from previous layer)
    run_layer(bufA, F, W2, as2, ad2, b2, HEADS, HID, 1);
    // Layer 3: [N,256] -> [N,32], heads=1, no ELU (mean over 1 head = identity)
    run_layer(bufA, F, W3, as3, ad3, b3, 1, HID, 0);

    // Global mean pool + classifier + log_softmax
    fill_k<<<gblk(GRAPHS * HID + GRAPHS, T), T, 0, stream>>>(sums, 0.f, GRAPHS * HID + GRAPHS);
    pool_k<<<gblk(Nn * HID, T), T, 0, stream>>>(bufA, batch, sums, cnt, Nn);
    classify_k<<<1, 64, 0, stream>>>(sums, cnt, Wc, bc, (float*)d_out);
}
